// ChildSumTreeLSTM_rel_10402410791855
// MI455X (gfx1250) — matrix operand order repaired and checked by reading, not compiled, not run on hardware
//
#include <hip/hip_runtime.h>
#include <hip/hip_bf16.h>
#include <math.h>

// ---------------------------------------------------------------------------
// ChildSum TreeLSTM for MI455X (gfx1250, wave32, WMMA).
// B=16 S=128 -> N=2048, H=DIN=256, E=2032, NLEV=21, L=12.
// GEMMs on v_wmma_f32_16x16x32_f16; A fragments via 2x b128 loads from
// row-major f16; B fragments pre-swizzled to per-lane contiguous 32B chunks.
// ---------------------------------------------------------------------------

typedef __attribute__((ext_vector_type(16))) _Float16 v16h;
typedef __attribute__((ext_vector_type(8)))  _Float16 v8h;
typedef __attribute__((ext_vector_type(8)))  float    v8f;

#define BB   16
#define SS   128
#define NNODE (BB*SS)      // 2048
#define HH   256
#define DINN 256
#define DE_  192
#define DR_  64
#define LL   12
#define NLEV_ 21
#define EPAD NNODE         // edge rows padded to 2048 for the edge GEMM
#define KK   256           // all GEMMs have K = 256

__device__ __forceinline__ float sigmoidf_(float x) {
    return 1.0f / (1.0f + expf(-x));
}

// ---------------------------------------------------------------------------
// f16 GEMM, fp32 accumulate:  C[M,N] = A[M,256] * B[256,N]
// One wave per 16x16 tile; K unrolled 8x32 via WMMA.
// A (row-major): lane covers row = lane%16; per 32-K block needs
//   k in [k0+16h, k0+16h+15]  (h = lane/16)
//   -> two contiguous v8h (b128) loads.
// B (pre-swizzled fragments): Bfrag[((kb*tilesN + nt)*32 + lane)*16 .. +15]
//   -> one contiguous v16h (2x b128) load.
// C/D: VGPR r -> M = r + 8*(lane/16), N = lane%16.
// ---------------------------------------------------------------------------
__global__ void gemm_f16_wmma(const _Float16* __restrict__ A,
                              const _Float16* __restrict__ Bfrag,
                              float* __restrict__ C,
                              int M, int N) {
    const int wave = (blockIdx.x * blockDim.x + threadIdx.x) >> 5;
    const int lane = threadIdx.x & 31;
    const int tilesN = N >> 4;
    const int tile_m = wave / tilesN;
    const int tile_n = wave - tile_m * tilesN;
    if (tile_m * 16 >= M) return;            // wave-uniform: EXEC stays all-1

    const int row0  = tile_m << 4;
    const int arow  = row0 + (lane & 15);
    const int ah8   = (lane >> 4) << 4;      // 0 or 16
    const _Float16* __restrict__ Arow  = A + (size_t)arow * KK;
    const _Float16* __restrict__ Bbase = Bfrag + ((size_t)tile_n * 32 + lane) * 16;
    const size_t bstride = (size_t)tilesN * 32 * 16;   // per 32-K block

    v8f acc = {};
#pragma unroll
    for (int kb = 0; kb < 8; ++kb) {
        const int k0 = kb << 5;
        const v8h alo = *(const v8h*)(Arow + k0 + ah8);
        const v8h ahi = *(const v8h*)(Arow + k0 + 8 + ah8);
        v16h a;
#pragma unroll
        for (int e = 0; e < 8; ++e) { a[e] = alo[e]; a[8 + e] = ahi[e]; }
        const v16h b = *(const v16h*)(Bbase + (size_t)kb * bstride);
        acc = __builtin_amdgcn_wmma_f32_16x16x32_f16(
            /*neg_a=*/false, a, /*neg_b=*/false, b,
            /*c_mod=*/(short)0, acc, /*reuse_a=*/false, /*reuse_b=*/false);
    }
    const int crow = row0 + ((lane >> 4) << 3);
    const int ccol = (tile_n << 4) + (lane & 15);
#pragma unroll
    for (int r = 0; r < 8; ++r)
        C[(size_t)(crow + r) * N + ccol] = acc[r];
}

// ---------------------------------------------------------------------------
// Setup kernels
// ---------------------------------------------------------------------------
__global__ void embed_kernel(const int* __restrict__ xs, const int* __restrict__ rels,
                             const float* __restrict__ emb_W, const float* __restrict__ rel_W,
                             _Float16* __restrict__ x_f16) {
    const int t = blockIdx.x * blockDim.x + threadIdx.x;   // n*256 + d
    const int n = t >> 8, d = t & 255;
    if (n >= NNODE) return;
    float v = (d < DE_) ? emb_W[xs[n] * DE_ + d]
                        : rel_W[rels[n] * DR_ + (d - DE_)];
    x_f16[t] = (_Float16)v;
}

__global__ void pack_weights_kernel(const float* __restrict__ W_ix, const float* __restrict__ W_fx,
                                    const float* __restrict__ W_ox, const float* __restrict__ W_ux,
                                    const float* __restrict__ W_ih, const float* __restrict__ W_oh,
                                    const float* __restrict__ W_uh, const float* __restrict__ W_fh,
                                    _Float16* __restrict__ Wx, _Float16* __restrict__ Wh,
                                    _Float16* __restrict__ Wfh) {
    const int t = blockIdx.x * blockDim.x + threadIdx.x;
    const int SZ_WX = 256 * 1024, SZ_WH = 256 * 768, SZ_WF = 256 * 256;
    if (t < SZ_WX) {
        const int k = t >> 10, j = t & 1023;
        const int q = j >> 8, col = j & 255;
        const float* src = (q == 0) ? W_ix : (q == 1) ? W_fx : (q == 2) ? W_ox : W_ux;
        Wx[t] = (_Float16)src[k * 256 + col];
    } else if (t < SZ_WX + SZ_WH) {
        const int u = t - SZ_WX;
        const int k = u / 768, j = u - k * 768;
        const int q = j >> 8, col = j & 255;
        const float* src = (q == 0) ? W_ih : (q == 1) ? W_oh : W_uh;
        Wh[u] = (_Float16)src[k * 256 + col];
    } else if (t < SZ_WX + SZ_WH + SZ_WF) {
        const int u = t - SZ_WX - SZ_WH;
        Wfh[u] = (_Float16)W_fh[u];
    }
}

// Row-major f16 [256, N] -> WMMA B-fragment order:
//   Bfrag[(((kb*tilesN)+nt)*32 + lane)*16 + e] = B[(kb*32 + (lane/16)*16 + e)*N + nt*16 + lane%16]
__global__ void swizzle_b_kernel(const _Float16* __restrict__ Bsrc,
                                 _Float16* __restrict__ Bfrag, int N) {
    const int t = blockIdx.x * blockDim.x + threadIdx.x;   // one lane-chunk each
    const int tilesN = N >> 4;
    const int total = 8 * tilesN * 32;                     // 8 k-blocks
    if (t >= total) return;
    const int lane = t & 31;
    const int nt   = (t >> 5) % tilesN;
    const int kb   = (t >> 5) / tilesN;
    const int col  = (nt << 4) + (lane & 15);
    const int krow = (kb << 5) + ((lane >> 4) << 4);
#pragma unroll
    for (int e = 0; e < 16; ++e)
        Bfrag[(size_t)t * 16 + e] = Bsrc[(size_t)(krow + e) * N + col];
}

__global__ void zero_kernel(float* __restrict__ p, int n) {
    const int t = blockIdx.x * blockDim.x + threadIdx.x;
    if (t < n) p[t] = 0.0f;
}

__global__ void f32_to_f16_kernel(const float* __restrict__ src,
                                  _Float16* __restrict__ dst, int n) {
    const int t = blockIdx.x * blockDim.x + threadIdx.x;
    if (t < n) dst[t] = (_Float16)src[t];
}

// ---------------------------------------------------------------------------
// Per-level kernels
// ---------------------------------------------------------------------------
__global__ void edge_scatter_kernel(const float* __restrict__ h,
                                    const int* __restrict__ child_idx,
                                    const int* __restrict__ parent_idx,
                                    int E,
                                    float* __restrict__ h_sum,
                                    _Float16* __restrict__ hch_f16) {
    const int t = blockIdx.x * blockDim.x + threadIdx.x;   // e*256 + d
    const int e = t >> 8, d = t & 255;
    if (e >= EPAD) return;
    if (e < E) {
        const int ch = child_idx[e];
        const int pa = parent_idx[e];
        const float hv = h[ch * HH + d];
        hch_f16[t] = (_Float16)hv;
        atomicAdd(&h_sum[pa * HH + d], hv);
    } else {
        hch_f16[t] = (_Float16)0.0f;   // pad rows so the edge GEMM is clean
    }
}

__global__ void fgate_scatter_kernel(const float* __restrict__ fpre,
                                     const float* __restrict__ xproj,
                                     const float* __restrict__ b_fx,
                                     const float* __restrict__ b_fh,
                                     const float* __restrict__ c,
                                     const int* __restrict__ child_idx,
                                     const int* __restrict__ parent_idx,
                                     int E,
                                     float* __restrict__ fc) {
    const int t = blockIdx.x * blockDim.x + threadIdx.x;
    const int e = t >> 8, d = t & 255;
    if (e >= E) return;
    const int pa = parent_idx[e];
    const int ch = child_idx[e];
    const float z = fpre[t] + b_fh[d] + b_fx[d] + xproj[pa * 1024 + 256 + d];
    const float f = sigmoidf_(z);
    atomicAdd(&fc[pa * HH + d], f * c[ch * HH + d]);
}

__global__ void commit_kernel(const float* __restrict__ xproj,
                              const float* __restrict__ g,
                              const float* __restrict__ b_ix,
                              const float* __restrict__ b_ih,
                              const int* __restrict__ node_height,
                              int level,
                              const float* __restrict__ fc,
                              float* __restrict__ h, float* __restrict__ c) {
    const int t = blockIdx.x * blockDim.x + threadIdx.x;   // n*256 + d
    const int n = t >> 8, d = t & 255;
    if (n >= NNODE) return;
    if (node_height[n] != level) return;
    const float i  = sigmoidf_(xproj[n * 1024 + d]        + b_ix[d] + g[n * 768 + d] + b_ih[d]);
    const float o  = sigmoidf_(xproj[n * 1024 + 512 + d]  + g[n * 768 + 256 + d]);
    const float u  = tanhf    (xproj[n * 1024 + 768 + d]  + g[n * 768 + 512 + d]);
    const float cn = i * u + fc[n * HH + d];
    const float hn = o * tanhf(cn);
    c[t] = cn;
    h[t] = hn;
}

// ---------------------------------------------------------------------------
// Max-pool over S then [16,256]x[256,12] output projection (one block / batch)
// ---------------------------------------------------------------------------
__global__ void pool_out_kernel(const float* __restrict__ h,
                                const float* __restrict__ W_out,
                                const float* __restrict__ b_out,
                                float* __restrict__ out) {
    __shared__ float pooled[HH];
    const int b = blockIdx.x, d = threadIdx.x;
    float m = -INFINITY;
    for (int s = 0; s < SS; ++s)
        m = fmaxf(m, h[(b * SS + s) * HH + d]);
    pooled[d] = m;
    __syncthreads();
    if (d < LL) {
        float acc = b_out[d];
        for (int k = 0; k < HH; ++k)
            acc += pooled[k] * W_out[k * LL + d];
        out[b * LL + d] = acc;
    }
}

// ---------------------------------------------------------------------------
// Host-side launch sequence
// ---------------------------------------------------------------------------
extern "C" void kernel_launch(void* const* d_in, const int* in_sizes, int n_in,
                              void* d_out, int out_size, void* d_ws, size_t ws_size,
                              hipStream_t stream) {
    const int*   xs          = (const int*)  d_in[0];
    const int*   rels        = (const int*)  d_in[1];
    const int*   child_idx   = (const int*)  d_in[2];
    const int*   parent_idx  = (const int*)  d_in[3];
    const int*   node_height = (const int*)  d_in[4];
    // d_in[5] = n_levels (compile-time NLEV_)
    const float* emb_W = (const float*)d_in[6];
    const float* rel_W = (const float*)d_in[7];
    const float* W_ix  = (const float*)d_in[8];
    const float* b_ix  = (const float*)d_in[9];
    const float* W_ih  = (const float*)d_in[10];
    const float* b_ih  = (const float*)d_in[11];
    const float* W_fx  = (const float*)d_in[12];
    const float* b_fx  = (const float*)d_in[13];
    const float* W_fh  = (const float*)d_in[14];
    const float* b_fh  = (const float*)d_in[15];
    const float* W_ox  = (const float*)d_in[16];
    const float* W_oh  = (const float*)d_in[17];
    const float* W_ux  = (const float*)d_in[18];
    const float* W_uh  = (const float*)d_in[19];
    const float* W_out = (const float*)d_in[20];
    const float* b_out = (const float*)d_in[21];
    const int E = in_sizes[2];   // 2032

    // ---- workspace layout (256B aligned) ----
    size_t off = 0;
    char* base = (char*)d_ws;
    auto alloc = [&](size_t bytes) -> void* {
        void* p = base + off;
        off += (bytes + 255) & ~(size_t)255;
        return p;
    };
    _Float16* x_f16    = (_Float16*)alloc((size_t)NNODE * DINN * 2);
    _Float16* Wx       = (_Float16*)alloc((size_t)KK * 1024 * 2);
    _Float16* Wh       = (_Float16*)alloc((size_t)KK * 768 * 2);
    _Float16* Wfh      = (_Float16*)alloc((size_t)KK * HH * 2);
    _Float16* WxF      = (_Float16*)alloc((size_t)KK * 1024 * 2);  // fragment order
    _Float16* WhF      = (_Float16*)alloc((size_t)KK * 768 * 2);
    _Float16* WfhF     = (_Float16*)alloc((size_t)KK * HH * 2);
    float*    xproj    = (float*)   alloc((size_t)NNODE * 1024 * 4);
    float*    h        = (float*)   alloc((size_t)NNODE * HH * 4);
    float*    c        = (float*)   alloc((size_t)NNODE * HH * 4);
    float*    h_sum    = (float*)   alloc((size_t)NNODE * HH * 4);
    _Float16* hsum_f16 = (_Float16*)alloc((size_t)NNODE * HH * 2);
    _Float16* hch_f16  = (_Float16*)alloc((size_t)EPAD * HH * 2);
    float*    fpre     = (float*)   alloc((size_t)EPAD * HH * 4);
    float*    g        = (float*)   alloc((size_t)NNODE * 768 * 4);
    float*    fc       = (float*)   alloc((size_t)NNODE * HH * 4);
    (void)ws_size;

    const int TB = 256;
    const int NH = NNODE * HH;                   // 524288

    // ---- setup ----
    embed_kernel<<<(NNODE * DINN) / TB, TB, 0, stream>>>(xs, rels, emb_W, rel_W, x_f16);
    {
        const int total = 256 * 1024 + 256 * 768 + 256 * 256;
        pack_weights_kernel<<<(total + TB - 1) / TB, TB, 0, stream>>>(
            W_ix, W_fx, W_ox, W_ux, W_ih, W_oh, W_uh, W_fh, Wx, Wh, Wfh);
    }
    swizzle_b_kernel<<<(8 * (1024 / 16) * 32 + TB - 1) / TB, TB, 0, stream>>>(Wx, WxF, 1024);
    swizzle_b_kernel<<<(8 * (768  / 16) * 32 + TB - 1) / TB, TB, 0, stream>>>(Wh, WhF, 768);
    swizzle_b_kernel<<<(8 * (256  / 16) * 32 + TB - 1) / TB, TB, 0, stream>>>(Wfh, WfhF, 256);

    zero_kernel<<<NH / TB, TB, 0, stream>>>(h, NH);
    zero_kernel<<<NH / TB, TB, 0, stream>>>(c, NH);

    // hoisted input projections: xproj[2048,1024] = x_f16 @ Wx   (WMMA)
    {
        const int waves = (NNODE / 16) * (1024 / 16);   // 8192
        gemm_f16_wmma<<<waves / 8, TB, 0, stream>>>(x_f16, WxF, xproj, NNODE, 1024);
    }

    // ---- 21 bottom-up levels ----
    for (int level = 0; level < NLEV_; ++level) {
        zero_kernel<<<NH / TB, TB, 0, stream>>>(h_sum, NH);
        zero_kernel<<<NH / TB, TB, 0, stream>>>(fc, NH);

        edge_scatter_kernel<<<(EPAD * HH) / TB, TB, 0, stream>>>(
            h, child_idx, parent_idx, E, h_sum, hch_f16);

        f32_to_f16_kernel<<<NH / TB, TB, 0, stream>>>(h_sum, hsum_f16, NH);

        // g[2048,768] = h_sum @ [W_ih|W_oh|W_uh]   (WMMA)
        {
            const int waves = (NNODE / 16) * (768 / 16);   // 6144
            gemm_f16_wmma<<<waves / 8, TB, 0, stream>>>(hsum_f16, WhF, g, NNODE, 768);
        }
        // fpre[2048,256] = h_ch @ W_fh   (WMMA, padded edge rows)
        {
            const int waves = (EPAD / 16) * (HH / 16);     // 2048
            gemm_f16_wmma<<<waves / 8, TB, 0, stream>>>(hch_f16, WfhF, fpre, EPAD, HH);
        }

        fgate_scatter_kernel<<<(EPAD * HH) / TB, TB, 0, stream>>>(
            fpre, xproj, b_fx, b_fh, c, child_idx, parent_idx, E, fc);

        commit_kernel<<<NH / TB, TB, 0, stream>>>(
            xproj, g, b_ix, b_ih, node_height, level, fc, h, c);
    }

    // ---- pool + output projection ----
    pool_out_kernel<<<BB, HH, 0, stream>>>(h, W_out, b_out, (float*)d_out);
}
